// LSTM_33809982554825
// MI455X (gfx1250) — matrix operand order repaired and checked by reading, not compiled, not run on hardware
//
#include <hip/hip_runtime.h>

// LSTM forward scan for MI455X (gfx1250, wave32).
// One persistent workgroup: 25 waves (800 threads), wave w owns gate rows
// [16w, 16w+16) of the 4H=400 recurrent pre-activations.
// Recurrent matvec z = W_hh * h via v_wmma_f32_16x16x32_f16:
//   K = 100 padded to 128 -> 4 WMMAs/wave/step in two independent 2-deep
//   accumulate chains (halves WMMA RAW stalls on the serial critical path).

typedef __attribute__((ext_vector_type(16))) _Float16 v16h;
typedef __attribute__((ext_vector_type(8)))  float    v8f;

#define T_SEQ   131072
#define ISZ     5
#define HSZ     100
#define KPAD    128            // K padded to 4 chunks of 32
#define NTILES  25             // 4*HSZ / 16
#define NTHREADS (NTILES * 32) // 800 threads = 25 wave32

__device__ __forceinline__ float fast_sigmoid(float x) {
    // v_exp_f32 + v_rcp_f32 instead of libm divide (v_div_fixup chain)
    return __builtin_amdgcn_rcpf(1.0f + __expf(-x));
}
__device__ __forceinline__ float fast_tanh(float x) {
    // tanh(x) = 1 - 2/(1 + e^{2x})
    return 1.0f - 2.0f * __builtin_amdgcn_rcpf(1.0f + __expf(2.0f * x));
}

__global__ __launch_bounds__(NTHREADS, 1)
void lstm_persistent_kernel(const float* __restrict__ input_seq, // [T, I]
                            const float* __restrict__ W_ih,      // [4H, I]
                            const float* __restrict__ W_hh,      // [4H, H]
                            const float* __restrict__ b_ih,      // [4H]
                            const float* __restrict__ b_hh,      // [4H]
                            const float* __restrict__ W_lin,     // [1, H]
                            const float* __restrict__ b_lin,     // [1]
                            float* __restrict__ out)             // [1]
{
    __shared__ __align__(32) _Float16 h16[KPAD]; // h in f16, zero-padded K>=100
    __shared__ float zraw[4 * HSZ];              // recurrent matvec result
    __shared__ float bsum[4 * HSZ];              // b_ih + b_hh
    __shared__ float wih_s[4 * HSZ * ISZ];       // W_ih staged in LDS
    __shared__ float h32[HSZ];                   // h in f32 (for final linear)
    __shared__ float xbuf[2][ISZ];               // double-buffered x_t

    const int tid  = threadIdx.x;
    const int wid  = tid >> 5;
    const int lane = tid & 31;

    // ---- one-time init ----
    for (int i = tid; i < KPAD; i += NTHREADS)          h16[i]  = (_Float16)0.0f;
    for (int i = tid; i < 4 * HSZ; i += NTHREADS)       bsum[i] = b_ih[i] + b_hh[i];
    for (int i = tid; i < 4 * HSZ * ISZ; i += NTHREADS) wih_s[i] = W_ih[i];
    for (int i = tid; i < HSZ; i += NTHREADS)           h32[i]  = 0.0f;
    if (tid < ISZ) xbuf[0][tid] = input_seq[tid];       // x(0)

    // A fragments: W_hh tile rows in f16, contiguous K order matching B.
    // lane<16 : element e -> K = base + e
    // lane>=16: element e -> K = base + 16 + e
    v16h afrag[4];
    {
        const int row = wid * 16 + (lane & 15);   // gate-output row, < 400
        #pragma unroll
        for (int ch = 0; ch < 4; ++ch) {
            #pragma unroll
            for (int e = 0; e < 16; ++e) {
                int k;
                if (lane < 16) k = ch * 32 + e;
                else           k = ch * 32 + e + 16;
                float w = (k < HSZ) ? W_hh[row * HSZ + k] : 0.0f;
                afrag[ch][e] = (_Float16)w;
            }
        }
    }

    float creg = 0.0f;          // cell state, lives in gate thread j = tid
    __syncthreads();

    // B in this code's contiguous K order, same as A (per 32-K chunk): lanes 0-15 hold K 0..15, lanes 16-31 hold K 16..31;
    // broadcast h chunk into all 16 columns -> every column of D equals z.
    const int boff = (lane & 16) ? 16 : 0;

    for (int t = 0; t < T_SEQ; ++t) {
        // ---- load all B chunks up front (one dscnt wait instead of four) ----
        v16h b0 = *(const v16h*)&h16[ 0 + boff];
        v16h b1 = *(const v16h*)&h16[32 + boff];
        v16h b2 = *(const v16h*)&h16[64 + boff];
        v16h b3 = *(const v16h*)&h16[96 + boff];

        // ---- two independent WMMA accumulate chains ----
        v8f acc0 = {};
        v8f acc1 = {};
        acc0 = __builtin_amdgcn_wmma_f32_16x16x32_f16(
                   false, afrag[0], false, b0, (short)0, acc0, false, false);
        acc1 = __builtin_amdgcn_wmma_f32_16x16x32_f16(
                   false, afrag[1], false, b1, (short)0, acc1, false, false);
        acc0 = __builtin_amdgcn_wmma_f32_16x16x32_f16(
                   false, afrag[2], false, b2, (short)0, acc0, false, false);
        acc1 = __builtin_amdgcn_wmma_f32_16x16x32_f16(
                   false, afrag[3], false, b3, (short)0, acc1, false, false);

        // D f32 16x16 layout: lane 0 holds z[tile+0..7] in acc[0..7],
        // lane 16 holds z[tile+8..15]. Combine the two chains only here.
        if (lane == 0 || lane == 16) {
            const int base = wid * 16 + (lane ? 8 : 0);
            #pragma unroll
            for (int m = 0; m < 8; ++m) zraw[base + m] = acc0[m] + acc1[m];
        }
        // prefetch x(t+1) a full step ahead
        if (tid < ISZ && (t + 1) < T_SEQ)
            xbuf[(t + 1) & 1][tid] = input_seq[(size_t)(t + 1) * ISZ + tid];
        __syncthreads();

        // ---- gate math: thread j owns hidden unit j ----
        if (tid < HSZ) {
            const int j = tid;
            const float x0 = xbuf[t & 1][0], x1 = xbuf[t & 1][1],
                        x2 = xbuf[t & 1][2], x3 = xbuf[t & 1][3],
                        x4 = xbuf[t & 1][4];
            float z[4];
            #pragma unroll
            for (int g = 0; g < 4; ++g) {   // gate order: i, f, g, o
                const int r = g * HSZ + j;
                const float* wr = &wih_s[r * ISZ];
                z[g] = zraw[r] + bsum[r]
                     + wr[0]*x0 + wr[1]*x1 + wr[2]*x2 + wr[3]*x3 + wr[4]*x4;
            }
            const float ig = fast_sigmoid(z[0]);
            const float fg = fast_sigmoid(z[1]);
            const float gg = fast_tanh(z[2]);
            const float og = fast_sigmoid(z[3]);
            creg = fg * creg + ig * gg;
            const float hv = og * fast_tanh(creg);
            h16[j] = (_Float16)hv;   // feeds next step's B operand
            h32[j] = hv;             // full precision for final linear
        }
        __syncthreads();
    }

    // ---- final linear: out = h_T @ W_lin.T + b_lin ----
    if (tid == 0) {
        float s = b_lin[0];
        for (int j = 0; j < HSZ; ++j) s += W_lin[j] * h32[j];
        out[0] = s;
    }
}

extern "C" void kernel_launch(void* const* d_in, const int* in_sizes, int n_in,
                              void* d_out, int out_size, void* d_ws, size_t ws_size,
                              hipStream_t stream) {
    (void)in_sizes; (void)n_in; (void)out_size; (void)d_ws; (void)ws_size;
    const float* input_seq = (const float*)d_in[0];
    const float* W_ih      = (const float*)d_in[1];
    const float* W_hh      = (const float*)d_in[2];
    const float* b_ih      = (const float*)d_in[3];
    const float* b_hh      = (const float*)d_in[4];
    const float* W_lin     = (const float*)d_in[5];
    const float* b_lin     = (const float*)d_in[6];
    lstm_persistent_kernel<<<dim3(1), dim3(NTHREADS), 0, stream>>>(
        input_seq, W_ih, W_hh, b_ih, b_hh, W_lin, b_lin, (float*)d_out);
}
